// OptimizedTransformerBlock_57655640982116
// MI455X (gfx1250) — matrix operand order repaired and checked by reading, not compiled, not run on hardware
//
#include <hip/hip_runtime.h>

typedef __attribute__((ext_vector_type(16))) __bf16 v16bf;
typedef __attribute__((ext_vector_type(8)))  float  v8f;

#define GEMM_BM 128
#define LSTR    34   // LDS row stride: 32 halfs + 1 pad dword (conflict-free)

#if defined(__gfx1250__) && __has_builtin(__builtin_amdgcn_tensor_load_to_lds) && \
    __has_builtin(__builtin_amdgcn_s_wait_tensorcnt)
#define USE_TDM 1
#else
#define USE_TDM 0
#endif

#if USE_TDM
typedef __attribute__((ext_vector_type(4))) unsigned v4u;
typedef __attribute__((ext_vector_type(8))) int      v8i;
typedef __attribute__((ext_vector_type(4))) int      v4i;

// 2D tile load via Tensor Data Mover (CDNA5 ISA).
// Fixed shape for this GEMM: d0 = 32 contiguous bf16, d1 rows, row stride
// `stride` elements, pad code 3 -> +1 DWORD every 16 DWORDs (34-half LDS rows).
// data_size=1 (2B), count=1, type=2; 2D -> remaining groups zero.
// 6-arg builtin on this toolchain: (u32x4, i32x8, i32x4, i32x4, i32x8, cpol).
__device__ __forceinline__ void tdm_load_tile(unsigned lds_off, const __bf16* gptr,
                                              unsigned d1, unsigned stride)
{
    const unsigned long long ga = (unsigned long long)(const void*)gptr;
    v4u g0;
    g0[0] = 1u;                                            // count=1, user descriptor
    g0[1] = lds_off;                                       // lds_addr (bytes)
    g0[2] = (unsigned)ga;                                  // global_addr[31:0]
    g0[3] = (unsigned)((ga >> 32) & 0x01ffffffu) | 0x80000000u; // addr[56:32] | type=2
    v8i g1;
    g1[0] = (int)((1u << 16) | (1u << 20) | (3u << 22));   // data_size=1, pad_en, pad_interval=3
    g1[1] = (int)(32u << 16);                              // tensor_dim0 = 32
    g1[2] = (int)((d1 & 0xffffu) << 16);                   // tensor_dim1[15:0]
    g1[3] = (int)((d1 >> 16) | (32u << 16));               // tensor_dim1[31:16] | tile_dim0=32
    g1[4] = (int)(d1 & 0xffffu);                           // tile_dim1 (tile_dim2=0)
    g1[5] = (int)stride;                                   // tensor_dim0_stride[31:0]
    g1[6] = 0;
    g1[7] = 0;
    v4i z4; z4[0] = z4[1] = z4[2] = z4[3] = 0;
    v8i z8;
    #pragma unroll
    for (int q = 0; q < 8; ++q) z8[q] = 0;
    __builtin_amdgcn_tensor_load_to_lds(g0, g1, z4, z4, z8, 0);
}

__device__ __forceinline__ unsigned lds_off_of(const void* p)
{
    return (unsigned)(unsigned long long)p;  // aperture low 32 bits = LDS offset
}
#endif

// ---------------------------------------------------------------------------
// bf16-WMMA GEMM computing C = A * B^T (+epilogue).
//   A: [M][K] bf16 row-major, leading dim lda
//   B: [N][K] bf16 row-major (K-major!), leading dim ldb
// Block tile 128 x BN, 256 threads = 8 wave32 waves (2x4); each wave owns
// 4 x (BN/64) 16x16 tiles; K stepped by 32 with V_WMMA_F32_16X16X32_BF16.
// Both tiles staged by TDM (double-buffered, s_wait_tensorcnt) into identical
// 34-half-row LDS layouts; fragments load as DWORD pairs for A and B alike.
// EPI: 0 Cf=acc | 1 Cb=acc | 2 Cf=aux+acc | 3 Cb=silu(aux)*acc
//      4 Cf += aux[m*8+expert]*acc
// Batch: z -> offsets (z>>4)*s?1 + (z&15)*s?2  (covers z = b*16 + h)
// ---------------------------------------------------------------------------
template<int BN, int EPI>
__global__ __launch_bounds__(256)
void gemm_wmma(const __bf16* __restrict__ A, int lda, long sA1, long sA2,
               const __bf16* __restrict__ B, int ldb, long sB1, long sB2,
               float* __restrict__ Cf, __bf16* __restrict__ Cb,
               int ldc, long sC1, long sC2,
               int K, const float* __restrict__ aux, int expert)
{
    __shared__ __bf16 lA[2][GEMM_BM * LSTR];
    __shared__ __bf16 lB[2][BN * LSTR];

    const int tid  = threadIdx.x;
    const int lane = tid & 31;
    const int wid  = tid >> 5;
    const int wr   = wid >> 2;
    const int wc   = wid & 3;
    constexpr int TN = BN / 64;

    const int m0 = blockIdx.y * GEMM_BM;
    const int n0 = blockIdx.x * BN;
    const int z  = blockIdx.z;
    const long offA = (long)(z >> 4) * sA1 + (long)(z & 15) * sA2;
    const long offB = (long)(z >> 4) * sB1 + (long)(z & 15) * sB2;
    const long offC = (long)(z >> 4) * sC1 + (long)(z & 15) * sC2;

    const int row = lane & 15;
    const int kh  = lane >> 4;

    const __bf16* Abase = A + offA + (long)m0 * lda;
    const __bf16* Bbase = B + offB + (long)n0 * ldb;

    v8f acc[4][TN];
    #pragma unroll
    for (int i = 0; i < 4; ++i)
        #pragma unroll
        for (int j = 0; j < TN; ++j)
            #pragma unroll
            for (int r = 0; r < 8; ++r) acc[i][j][r] = 0.0f;

#if USE_TDM
    auto issue = [&](int kt, int st) {
        tdm_load_tile(lds_off_of(&lA[st][0]), Abase + kt, (unsigned)GEMM_BM, (unsigned)lda);
        tdm_load_tile(lds_off_of(&lB[st][0]), Bbase + kt, (unsigned)BN,      (unsigned)ldb);
    };
    if (wid == 0) issue(0, 0);
#endif

    for (int kt = 0, it = 0; kt < K; kt += 32, ++it) {
        const int st = it & 1;
#if USE_TDM
        if (wid == 0) {
            if (kt + 32 < K) {
                issue(kt + 32, st ^ 1);
                __builtin_amdgcn_s_wait_tensorcnt(2);   // current stage's pair retired
            } else {
                __builtin_amdgcn_s_wait_tensorcnt(0);
            }
        }
        __syncthreads();
#else
        {
            const unsigned* Ag = (const unsigned*)Abase;
            for (int t2 = tid; t2 < GEMM_BM * 16; t2 += 256) {
                const int r = t2 >> 4, c2 = t2 & 15;
                *(unsigned*)&lA[st][r * LSTR + c2 * 2] =
                    Ag[(((long)r * lda + kt) >> 1) + c2];
            }
            const unsigned* Bg = (const unsigned*)Bbase;
            for (int t2 = tid; t2 < BN * 16; t2 += 256) {
                const int n = t2 >> 4, c2 = t2 & 15;
                *(unsigned*)&lB[st][n * LSTR + c2 * 2] =
                    Bg[(((long)n * ldb + kt) >> 1) + c2];
            }
        }
        __syncthreads();
#endif
        // ---- fragments (ISA VGPR layouts) + WMMA ----
        const __bf16* lAs = lA[st];
        const __bf16* lBs = lB[st];
        v16bf af[4];
        #pragma unroll
        for (int i = 0; i < 4; ++i) {
            const __bf16* ap = lAs + (wr * 64 + i * 16 + row) * LSTR;
            #pragma unroll
            for (int v = 0; v < 8; ++v) {
                const int k0 = (v < 4) ? (kh * 8 + 2 * v) : (16 + kh * 8 + 2 * (v - 4));
                af[i][2 * v]     = ap[k0];
                af[i][2 * v + 1] = ap[k0 + 1];
            }
        }
        #pragma unroll
        for (int j = 0; j < TN; ++j) {
            const __bf16* bp = lBs + (wc * (BN / 4) + j * 16 + row) * LSTR;
            v16bf bfrag;
            #pragma unroll
            for (int v = 0; v < 8; ++v) {
                const int k0 = (v < 4) ? (kh * 8 + 2 * v) : (16 + kh * 8 + 2 * (v - 4));
                bfrag[2 * v]     = bp[k0];
                bfrag[2 * v + 1] = bp[k0 + 1];
            }
            #pragma unroll
            for (int i = 0; i < 4; ++i)
                acc[i][j] = __builtin_amdgcn_wmma_f32_16x16x32_bf16(
                    false, af[i], false, bfrag, (short)0, acc[i][j], false, false);
        }
        __syncthreads();
    }

    // ---- epilogue: one base per 16x16 tile, pointer-step by ldc ----
    #pragma unroll
    for (int i = 0; i < 4; ++i)
        #pragma unroll
        for (int j = 0; j < TN; ++j) {
            const int mb = m0 + wr * 64 + i * 16 + kh * 8;
            const int n  = n0 + wc * (BN / 4) + j * 16 + row;
            const long cbase = offC + (long)mb * ldc + n;
            if constexpr (EPI == 0) {
                float* p = Cf + cbase;
                #pragma unroll
                for (int r = 0; r < 8; ++r) { *p = acc[i][j][r]; p += ldc; }
            } else if constexpr (EPI == 1) {
                __bf16* p = Cb + cbase;
                #pragma unroll
                for (int r = 0; r < 8; ++r) { *p = (__bf16)acc[i][j][r]; p += ldc; }
            } else if constexpr (EPI == 2) {
                float* p = Cf + cbase;
                const float* q = aux + cbase;
                #pragma unroll
                for (int r = 0; r < 8; ++r) { *p = *q + acc[i][j][r]; p += ldc; q += ldc; }
            } else if constexpr (EPI == 3) {
                __bf16* p = Cb + cbase;
                const float* q = aux + cbase;
                #pragma unroll
                for (int r = 0; r < 8; ++r) {
                    const float g = *q;
                    *p = (__bf16)(acc[i][j][r] * (g / (1.0f + expf(-g))));
                    p += ldc; q += ldc;
                }
            } else {
                float* p = Cf + cbase;
                const float* q = aux + (long)mb * 8 + expert;
                #pragma unroll
                for (int r = 0; r < 8; ++r) { *p += *q * acc[i][j][r]; p += ldc; q += 8; }
            }
        }
}

// ---------------------------------------------------------------------------
// Transposing f32 -> bf16 conversion: in f32 [R][C] -> out bf16 [C][R].
// 32x32 tiles through LDS; coalesced reads and writes. blockIdx.z batches
// matrices of R*C elements (for the 8 expert weight tensors).
// ---------------------------------------------------------------------------
__global__ __launch_bounds__(256)
void cvt_t_bf16_kernel(const float* __restrict__ in, __bf16* __restrict__ out,
                       int R, int C)
{
    __shared__ __bf16 tile[32][33];
    const long mo = (long)blockIdx.z * (long)R * C;
    const int r0 = blockIdx.y * 32, c0 = blockIdx.x * 32;
    const int tx = threadIdx.x & 31;
    const int ty = threadIdx.x >> 5;          // 8 groups of 4 rows
    #pragma unroll
    for (int j = 0; j < 4; ++j) {
        const int r = ty * 4 + j;
        tile[tx][r] = (__bf16)in[mo + (long)(r0 + r) * C + c0 + tx];
    }
    __syncthreads();
    #pragma unroll
    for (int j = 0; j < 4; ++j) {
        const int c = ty * 4 + j;
        out[mo + (long)(c0 + c) * R + r0 + tx] = tile[c][tx];
    }
}

// ---------------------------------------------------------------------------
// LayerNorm over D=1024, one block per token; optional raw-row copy to seed
// d_out with the residual stream before the MoE accumulation.
// ---------------------------------------------------------------------------
__global__ __launch_bounds__(256)
void ln_kernel(const float* __restrict__ x, const float* __restrict__ g,
               const float* __restrict__ b, __bf16* __restrict__ outb,
               float* __restrict__ copyout)
{
    __shared__ float red[256];
    const long base = (long)blockIdx.x * 1024;
    const int tid = threadIdx.x;
    float v[4];
    float s = 0.f;
    #pragma unroll
    for (int j = 0; j < 4; ++j) { v[j] = x[base + tid + j * 256]; s += v[j]; }
    red[tid] = s; __syncthreads();
    for (int o = 128; o > 0; o >>= 1) { if (tid < o) red[tid] += red[tid + o]; __syncthreads(); }
    const float mu = red[0] * (1.0f / 1024.0f);
    __syncthreads();
    float s2 = 0.f;
    #pragma unroll
    for (int j = 0; j < 4; ++j) { const float d = v[j] - mu; s2 += d * d; }
    red[tid] = s2; __syncthreads();
    for (int o = 128; o > 0; o >>= 1) { if (tid < o) red[tid] += red[tid + o]; __syncthreads(); }
    const float rs = rsqrtf(red[0] * (1.0f / 1024.0f) + 1e-5f);
    #pragma unroll
    for (int j = 0; j < 4; ++j) {
        const int d = tid + j * 256;
        outb[base + d] = (__bf16)((v[j] - mu) * rs * g[d] + b[d]);
        if (copyout) copyout[base + d] = v[j];
    }
}

// ---------------------------------------------------------------------------
// RoPE on q,k + bf16 conversion. q,k written [b,s,h,hd] (K-major per head row);
// v written TRANSPOSED per head: vbt[((b*16+h)*64 + d)*512 + s] so the P*V
// GEMM's B operand is K-major too.
// ---------------------------------------------------------------------------
__global__ __launch_bounds__(256)
void rope_kernel(const float* __restrict__ qkv, __bf16* __restrict__ qb,
                 __bf16* __restrict__ kb, __bf16* __restrict__ vbt)
{
    const int gid = blockIdx.x * 256 + threadIdx.x;   // T*16*32 threads
    const int i = gid & 31;
    const int h = (gid >> 5) & 15;
    const int t = gid >> 9;
    const int s = t & 511;
    const int bidx = t >> 9;
    const float inv = expf(-(float)(2 * i) * (9.210340371976184f / 64.0f)); // 10000^(-2i/64)
    const float fr = (float)s * inv;
    const float c = cosf(fr), sn = sinf(fr);
    const float* base = qkv + (long)t * 3072 + h * 64 + 2 * i;
    const float qe = base[0],    qo = base[1];
    const float ke = base[1024], ko = base[1025];
    const float ve = base[2048], vo = base[2049];
    const long ob = (long)t * 1024 + h * 64 + 2 * i;
    qb[ob]     = (__bf16)(qe * c - qo * sn);
    qb[ob + 1] = (__bf16)(qe * sn + qo * c);
    kb[ob]     = (__bf16)(ke * c - ko * sn);
    kb[ob + 1] = (__bf16)(ke * sn + ko * c);
    const long vt = ((long)(bidx * 16 + h) * 64 + 2 * i) * 512 + s;
    vbt[vt]       = (__bf16)ve;
    vbt[vt + 512] = (__bf16)vo;
}

// ---------------------------------------------------------------------------
// Causal softmax over one 512-wide score row per block; writes bf16 probs.
// ---------------------------------------------------------------------------
__global__ __launch_bounds__(256)
void softmax_kernel(const float* __restrict__ scores, __bf16* __restrict__ attnb)
{
    __shared__ float red[256];
    const long rowid = blockIdx.x;
    const int  sq  = (int)(rowid & 511);
    const int  tid = threadIdx.x;
    const float* p = scores + rowid * 512;
    const float v0 = (tid       <= sq) ? p[tid]       * 0.125f : -3.0e38f;
    const float v1 = (tid + 256 <= sq) ? p[tid + 256] * 0.125f : -3.0e38f;
    red[tid] = fmaxf(v0, v1); __syncthreads();
    for (int o = 128; o > 0; o >>= 1) { if (tid < o) red[tid] = fmaxf(red[tid], red[tid + o]); __syncthreads(); }
    const float m = red[0]; __syncthreads();
    const float e0 = (tid       <= sq) ? expf(v0 - m) : 0.f;
    const float e1 = (tid + 256 <= sq) ? expf(v1 - m) : 0.f;
    red[tid] = e0 + e1; __syncthreads();
    for (int o = 128; o > 0; o >>= 1) { if (tid < o) red[tid] += red[tid + o]; __syncthreads(); }
    const float inv = 1.0f / red[0];
    attnb[rowid * 512 + tid]       = (__bf16)(e0 * inv);
    attnb[rowid * 512 + tid + 256] = (__bf16)(e1 * inv);
}

// ---------------------------------------------------------------------------
// Router: one wave per token. bf16 gate matmul (K=1024, E=8), full softmax for
// expert_dist (atomic accumulate), top-2 + softmax -> combine weights cw.
// ---------------------------------------------------------------------------
__global__ __launch_bounds__(256)
void gate_kernel(const __bf16* __restrict__ xn2, const float* __restrict__ gw,
                 float* __restrict__ cw, float* __restrict__ dist)
{
    const int wid  = threadIdx.x >> 5;
    const int lane = threadIdx.x & 31;
    const int t = blockIdx.x * 8 + wid;
    float acc[8];
    #pragma unroll
    for (int e = 0; e < 8; ++e) acc[e] = 0.f;
    const __bf16* xp = xn2 + (long)t * 1024;
    for (int d = lane; d < 1024; d += 32) {
        const float xv = (float)xp[d];
        const float* gp = gw + (long)d * 8;
        #pragma unroll
        for (int e = 0; e < 8; ++e) acc[e] += xv * (float)(__bf16)gp[e];
    }
    #pragma unroll
    for (int e = 0; e < 8; ++e)
        for (int o = 16; o > 0; o >>= 1) acc[e] += __shfl_xor(acc[e], o, 32);
    if (lane == 0) {
        float m = acc[0];
        for (int e = 1; e < 8; ++e) m = fmaxf(m, acc[e]);
        float p[8], se = 0.f;
        for (int e = 0; e < 8; ++e) { p[e] = expf(acc[e] - m); se += p[e]; }
        const float invse = 1.0f / se;
        for (int e = 0; e < 8; ++e) atomicAdd(&dist[e], p[e] * invse * (1.0f / 2048.0f));
        int i0 = 0;
        for (int e = 1; e < 8; ++e) if (acc[e] > acc[i0]) i0 = e;
        int i1 = (i0 == 0) ? 1 : 0;
        for (int e = 0; e < 8; ++e) if (e != i0 && acc[e] > acc[i1]) i1 = e;
        const float e1 = expf(acc[i1] - acc[i0]);
        const float w0 = 1.0f / (1.0f + e1);
        float* cwp = cw + (long)t * 8;
        for (int e = 0; e < 8; ++e) cwp[e] = 0.f;
        cwp[i0] = w0;
        cwp[i1] = 1.0f - w0;
    }
}

__global__ void zero_dist(float* dist) { if (threadIdx.x < 8) dist[threadIdx.x] = 0.f; }

__global__ void finalize_kernel(const float* __restrict__ dist, float* __restrict__ tail)
{
    if (threadIdx.x == 0) {
        float s = 0.f;
        for (int e = 0; e < 8; ++e) s += dist[e] * dist[e];
        tail[0] = 8.0f * s;                       // lb_loss
        for (int e = 0; e < 8; ++e) tail[1 + e] = dist[e];
    }
}

// ---------------------------------------------------------------------------
extern "C" void kernel_launch(void* const* d_in, const int* in_sizes, int n_in,
                              void* d_out, int out_size, void* d_ws, size_t ws_size,
                              hipStream_t stream)
{
    const float* x      = (const float*)d_in[0];
    const float* ln1_g  = (const float*)d_in[1];
    const float* ln1_b  = (const float*)d_in[2];
    const float* ln2_g  = (const float*)d_in[3];
    const float* ln2_b  = (const float*)d_in[4];
    const float* w_qkv  = (const float*)d_in[5];
    const float* w_o    = (const float*)d_in[6];
    const float* gate_w = (const float*)d_in[7];
    const float* w_gate = (const float*)d_in[8];
    const float* w_up   = (const float*)d_in[9];
    const float* w_down = (const float*)d_in[10];
    float* out = (float*)d_out;

    constexpr long T = 2048, D = 1024, S = 512, Hd = 1792;
    constexpr long EWN = 8L * D * Hd;          // elements in each expert weight set

    char* wsb = (char*)d_ws;
    size_t off = 0;
    auto alloc = [&](size_t bytes) -> char* {
        char* p = wsb + off;
        off = (off + bytes + 255) & ~(size_t)255;
        return p;
    };
    __bf16* xn   = (__bf16*)alloc(T * D * 2);
    __bf16* qb   = (__bf16*)alloc(T * D * 2);
    __bf16* kb   = (__bf16*)alloc(T * D * 2);
    __bf16* vbt  = (__bf16*)alloc(T * D * 2);      // V transposed per head [b,h,hd,S]
    __bf16* ao   = (__bf16*)alloc(T * D * 2);
    float*  x1   = (float*) alloc(T * D * 4);
    __bf16* xn2  = (__bf16*)alloc(T * D * 2);
    __bf16* wqkvT= (__bf16*)alloc(D * 3 * D * 2);  // [3072][1024]
    __bf16* woT  = (__bf16*)alloc(D * D * 2);      // [1024][1024]
    float*  cw   = (float*) alloc(T * 8 * 4);
    float*  dist = (float*) alloc(64);
    // Region 1 (24 MB): qkv f32 during attention; gbuf+hb during MoE.
    char*   R1   = alloc(T * 3 * D * 4);
    float*  qkv  = (float*)R1;
    float*  gbuf = (float*)R1;
    __bf16* hb   = (__bf16*)(R1 + T * Hd * 4);
    // Region 2 (96 MB): scores f32 + probs bf16 during attention; transposed
    // bf16 expert weights (3 x 28 MB) during MoE (converted after attention).
    char*   R2   = alloc(64L * S * S * 4 + 64L * S * S * 2);
    float*  sc   = (float*)R2;
    __bf16* at   = (__bf16*)(R2 + 64L * S * S * 4);
    __bf16* wgT  = (__bf16*)R2;                    // 8 x [1792][1024]
    __bf16* wuT  = (__bf16*)(R2 + EWN * 2);        // 8 x [1792][1024]
    __bf16* wdT  = (__bf16*)(R2 + EWN * 4);        // 8 x [1024][1792]

    // 1) attention weight transpose-conversions + LN1
    cvt_t_bf16_kernel<<<dim3(3 * D / 32, D / 32, 1), 256, 0, stream>>>(w_qkv, wqkvT, (int)D, (int)(3 * D));
    cvt_t_bf16_kernel<<<dim3(D / 32, D / 32, 1), 256, 0, stream>>>(w_o, woT, (int)D, (int)D);
    ln_kernel<<<(unsigned)T, 256, 0, stream>>>(x, ln1_g, ln1_b, xn, nullptr);
    // 2) QKV projection: xn [2048][1024] * wqkvT^T
    gemm_wmma<128, 0><<<dim3(3 * D / 128, T / 128, 1), 256, 0, stream>>>(
        xn, (int)D, 0, 0, wqkvT, (int)D, 0, 0,
        qkv, nullptr, (int)(3 * D), 0, 0, (int)D, nullptr, 0);
    // 3) RoPE + bf16 q/k, transposed v
    rope_kernel<<<(unsigned)(T * 512 / 256), 256, 0, stream>>>(qkv, qb, kb, vbt);
    // 4) scores = Q * K^T per (b,h)  (kb rows are K-major per head already)
    gemm_wmma<128, 0><<<dim3(S / 128, S / 128, 64), 256, 0, stream>>>(
        qb, (int)D, S * D, 64, kb, (int)D, S * D, 64,
        sc, nullptr, (int)S, 16L * S * S, (long)S * S, 64, nullptr, 0);
    // 5) causal softmax -> bf16 probs
    softmax_kernel<<<(unsigned)(64L * S), 256, 0, stream>>>(sc, at);
    // 6) attn_out = P * V per (b,h): B = vbt [64][512] per head (K-major)
    gemm_wmma<64, 1><<<dim3(1, S / 128, 64), 256, 0, stream>>>(
        at, (int)S, 16L * S * S, (long)S * S, vbt, (int)S, 16L * 64 * S, 64L * S,
        nullptr, ao, (int)D, S * D, 64, (int)S, nullptr, 0);
    // 7) x1 = x + attn_out @ w_o
    gemm_wmma<128, 2><<<dim3(D / 128, T / 128, 1), 256, 0, stream>>>(
        ao, (int)D, 0, 0, woT, (int)D, 0, 0,
        x1, nullptr, (int)D, 0, 0, (int)D, x, 0);
    // 8) LN2 -> bf16, seed d_out with x1 (residual stream)
    ln_kernel<<<(unsigned)T, 256, 0, stream>>>(x1, ln2_g, ln2_b, xn2, out);
    // 9) transpose-convert expert weights into R2 (scores/probs now dead)
    cvt_t_bf16_kernel<<<dim3(Hd / 32, D / 32, 8), 256, 0, stream>>>(w_gate, wgT, (int)D, (int)Hd);
    cvt_t_bf16_kernel<<<dim3(Hd / 32, D / 32, 8), 256, 0, stream>>>(w_up,   wuT, (int)D, (int)Hd);
    cvt_t_bf16_kernel<<<dim3(D / 32, Hd / 32, 8), 256, 0, stream>>>(w_down, wdT, (int)Hd, (int)D);
    // 10) router stats + combine weights
    zero_dist<<<1, 32, 0, stream>>>(dist);
    gate_kernel<<<(unsigned)(T / 8), 256, 0, stream>>>(xn2, gate_w, cw, dist);
    // 11) dense experts: g, h = silu(g)*u, d_out += cw[:,e] * (h @ w_down[e])
    for (int e = 0; e < 8; ++e) {
        const __bf16* wg = wgT + (long)e * D * Hd;
        const __bf16* wu = wuT + (long)e * D * Hd;
        const __bf16* wd = wdT + (long)e * Hd * D;
        gemm_wmma<128, 0><<<dim3(Hd / 128, T / 128, 1), 256, 0, stream>>>(
            xn2, (int)D, 0, 0, wg, (int)D, 0, 0,
            gbuf, nullptr, (int)Hd, 0, 0, (int)D, nullptr, 0);
        gemm_wmma<128, 3><<<dim3(Hd / 128, T / 128, 1), 256, 0, stream>>>(
            xn2, (int)D, 0, 0, wu, (int)D, 0, 0,
            nullptr, hb, (int)Hd, 0, 0, (int)D, gbuf, 0);
        gemm_wmma<128, 4><<<dim3(D / 128, T / 128, 1), 256, 0, stream>>>(
            hb, (int)Hd, 0, 0, wd, (int)Hd, 0, 0,
            out, nullptr, (int)D, 0, 0, (int)Hd, cw, e);
    }
    // 12) lb_loss + expert_dist tail
    finalize_kernel<<<1, 32, 0, stream>>>(dist, out + T * D);
}
